// VOXCPM_MAIN_37151467110418
// MI455X (gfx1250) — compile-verified, matrix operand order checked by reading, not run on hardware
//
#include <hip/hip_runtime.h>
#include <hip/hip_bf16.h>
#include <math.h>

// ---- problem constants (from reference) ----
#define LL    8
#define HH    1024
#define NHD   16
#define KVHD  2
#define HDM   64
#define DFFD  4096
#define GG    (NHD / KVHD)      // 8
#define SS    256
#define PASTT 2048
#define TT    (SS + PASTT)      // 2304
#define QKHD  (NHD + KVHD)      // 18
#define QKVN  ((NHD + 2 * KVHD) * HDM)  // 1280

typedef __attribute__((ext_vector_type(16))) _Float16 v16h;
typedef __attribute__((ext_vector_type(8)))  float    v8f;

__device__ __forceinline__ v8f wmma_f16(v16h a, v16h b, v8f c) {
  return __builtin_amdgcn_wmma_f32_16x16x32_f16(
      /*neg_a=*/false, a, /*neg_b=*/false, b,
      /*c_mod=*/(short)0, c, /*reuse_a=*/false, /*reuse_b=*/false);
}

// ---------------------------------------------------------------------------
// WMMA fragment gathers. 16-bit A/B layout (CDNA5 ISA 7.12.2): lane holds
// row (A: m=lane&15) / col (B: n=lane&15); with sub=(lane&16)?8:0 the halves
// are  [0..7] = K[k0+sub .. k0+sub+7],  [8..15] = K[k0+16+sub .. k0+23+sub].
// All gathers below are 16-byte-contiguous segments.
// ---------------------------------------------------------------------------
__device__ __forceinline__ v16h frag_from_f32(const float* __restrict__ row,
                                              int k0, int lane) {
  const int sub = (lane & 16) ? 8 : 0;
  const float* p0 = row + k0 + sub;
  const float* p1 = row + k0 + 16 + sub;
  v16h f;
#pragma unroll
  for (int i = 0; i < 8; ++i) f[i] = (_Float16)p0[i];
#pragma unroll
  for (int i = 0; i < 8; ++i) f[i + 8] = (_Float16)p1[i];
  return f;
}

__device__ __forceinline__ v16h frag_from_f16(const _Float16* __restrict__ row,
                                              int k0, int lane) {
  const int sub = (lane & 16) ? 8 : 0;
  const _Float16* p0 = row + k0 + sub;
  const _Float16* p1 = row + k0 + 16 + sub;
  v16h f;
#pragma unroll
  for (int i = 0; i < 8; ++i) f[i] = p0[i];
#pragma unroll
  for (int i = 0; i < 8; ++i) f[i + 8] = p1[i];
  return f;
}

// ---------------------------------------------------------------------------
// C(256 x N) = A(256 x K) @ W(N x K)^T (+ optional residual, templated).
// Block = 128 threads = 4 waves; block tile = 16(M) x 128(N); each wave owns
// a 16x32 C tile (2 accumulators). A tile (16x64 f16) staged in LDS and
// shared by all 4 waves; 4 v_wmma per K-loop iteration per wave.
// ---------------------------------------------------------------------------
template <bool HAS_RES>
__global__ void __launch_bounds__(128)
gemm_xwT_wmma(const float* __restrict__ A, const float* __restrict__ W,
              float* __restrict__ C, const float* __restrict__ Res,
              int N, int K) {
  __shared__ __align__(16) _Float16 As[16 * 64];
  const int lane = threadIdx.x & 31;
  const int wave = threadIdx.x >> 5;
  const int m0 = blockIdx.x << 4;
  const int n0 = (blockIdx.y << 7) + (wave << 5);
  const int r  = lane & 15;

  const float* __restrict__ w0 = W + (size_t)(n0 + r) * K;
  const float* __restrict__ w1 = W + (size_t)(n0 + 16 + r) * K;

  v8f acc0 = {}, acc1 = {};

  for (int k0 = 0; k0 < K; k0 += 64) {
    __syncthreads();
#pragma unroll
    for (int i = threadIdx.x; i < 16 * 64; i += 128)
      As[i] = (_Float16)A[(size_t)(m0 + (i >> 6)) * K + k0 + (i & 63)];
    __syncthreads();

    __builtin_prefetch(w0 + k0 + 64, 0, 1);
    __builtin_prefetch(w1 + k0 + 64, 0, 1);
#pragma unroll
    for (int kk = 0; kk < 64; kk += 32) {
      const v16h a  = frag_from_f16(As + r * 64, kk, lane);
      const v16h b0 = frag_from_f32(w0, k0 + kk, lane);
      const v16h b1 = frag_from_f32(w1, k0 + kk, lane);
      acc0 = wmma_f16(a, b0, acc0);
      acc1 = wmma_f16(a, b1, acc1);
    }
  }

  const int mbase = m0 + ((lane >> 4) << 3);
#pragma unroll
  for (int i = 0; i < 8; ++i) {
    const size_t off = (size_t)(mbase + i) * N + n0 + r;
    float v0 = acc0[i], v1 = acc1[i];
    if (HAS_RES) { v0 += Res[off]; v1 += Res[off + 16]; }
    C[off]      = v0;
    C[off + 16] = v1;
  }
}

// ---------------------------------------------------------------------------
// x_norm = h * rsqrt(sum(h^2))      (reference _rms_sum: no mean, no eps)
// ---------------------------------------------------------------------------
__global__ void __launch_bounds__(256)
rmsnorm_sum_kernel(const float* __restrict__ h, float* __restrict__ xn) {
  __shared__ float red[256];
  const int row = blockIdx.x;
  const float* x = h + (size_t)row * HH;
  float s = 0.f;
  for (int i = threadIdx.x; i < HH; i += 256) { float v = x[i]; s += v * v; }
  red[threadIdx.x] = s; __syncthreads();
  for (int off = 128; off > 0; off >>= 1) {
    if (threadIdx.x < off) red[threadIdx.x] += red[threadIdx.x + off];
    __syncthreads();
  }
  const float inv = rsqrtf(red[0]);
  for (int i = threadIdx.x; i < HH; i += 256)
    xn[(size_t)row * HH + i] = x[i] * inv;
}

// out = h * rsqrt(mean(h^2) + 1e-6) * norm_w
__global__ void __launch_bounds__(256)
final_norm_kernel(const float* __restrict__ h, const float* __restrict__ nw,
                  float* __restrict__ out) {
  __shared__ float red[256];
  const int row = blockIdx.x;
  const float* x = h + (size_t)row * HH;
  float s = 0.f;
  for (int i = threadIdx.x; i < HH; i += 256) { float v = x[i]; s += v * v; }
  red[threadIdx.x] = s; __syncthreads();
  for (int off = 128; off > 0; off >>= 1) {
    if (threadIdx.x < off) red[threadIdx.x] += red[threadIdx.x + off];
    __syncthreads();
  }
  const float inv = rsqrtf(red[0] / (float)HH + 1e-6f);
  for (int i = threadIdx.x; i < HH; i += 256)
    out[(size_t)row * HH + i] = x[i] * inv * nw[i];
}

// ---------------------------------------------------------------------------
// RoPE (reference form): out[d] = x[d]*cos[d] + x[(d+32)%64]*sin[d],
// in place on the first QKH heads. One thread per (s, head, d<32).
// ---------------------------------------------------------------------------
__global__ void rope_qk_kernel(float* __restrict__ qkv,
                               const float* __restrict__ cosb,
                               const float* __restrict__ sinb) {
  int idx = blockIdx.x * blockDim.x + threadIdx.x;
  if (idx >= SS * QKHD * 32) return;
  const int d    = idx & 31;
  const int head = (idx >> 5) % QKHD;
  const int s    = idx / (QKHD * 32);
  float* p = qkv + (size_t)s * QKVN + head * HDM;
  const float x0 = p[d], x1 = p[d + 32];
  const float c0 = cosb[s * HDM + d],      s0 = sinb[s * HDM + d];
  const float c1 = cosb[s * HDM + d + 32], s1 = sinb[s * HDM + d + 32];
  p[d]      = x0 * c0 + x1 * s0;
  p[d + 32] = x1 * c1 + x0 * s1;
}

// K cache, f16, layout [kvh][t][d]  (WMMA-B fragment rows = keys)
__global__ void build_K_kernel(const float* __restrict__ qkv,
                               const float* __restrict__ pastk,
                               _Float16* __restrict__ Kf) {
  int idx = blockIdx.x * blockDim.x + threadIdx.x;
  if (idx >= KVHD * TT * HDM) return;
  const int d   = idx % HDM;
  const int t   = (idx / HDM) % TT;
  const int kvh = idx / (HDM * TT);
  float v;
  if (t < PASTT) v = pastk[((size_t)kvh * HDM + d) * PASTT + t];
  else           v = qkv[(size_t)(t - PASTT) * QKVN + (NHD + kvh) * HDM + d];
  Kf[idx] = (_Float16)v;
}

// V cache transposed, f16, layout [kvh][d][t]  (WMMA-B fragment rows = dims)
__global__ void build_V_kernel(const float* __restrict__ qkv,
                               const float* __restrict__ pastv,
                               _Float16* __restrict__ Vf) {
  int idx = blockIdx.x * blockDim.x + threadIdx.x;
  if (idx >= KVHD * HDM * TT) return;
  const int t   = idx % TT;
  const int d   = (idx / TT) % HDM;
  const int kvh = idx / (TT * HDM);
  float v;
  if (t < PASTT) v = pastv[((size_t)kvh * PASTT + t) * HDM + d];
  else           v = qkv[(size_t)(t - PASTT) * QKVN + (QKHD + kvh) * HDM + d];
  Vf[idx] = (_Float16)v;
}

// ---------------------------------------------------------------------------
// Flash-style WMMA attention. Block = 4 waves; wave owns one 16-row Q tile
// of one head. Streams T in chunks of 32: 4 v_wmma for S = Q@K^T, online
// softmax via 16-lane shfl reductions (C-layout rows are elementwise in the
// accumulator index, so rescale is per (lane, slot)), P re-laid out C->A
// through a private LDS slice, then 4 v_wmma for O += P@V.
// No softmax scale (reference has none); causal mask inline (== mask input).
// ---------------------------------------------------------------------------
__global__ void __launch_bounds__(128)
attn_wmma_kernel(const float* __restrict__ qkv, const _Float16* __restrict__ Kf,
                 const _Float16* __restrict__ Vf, float* __restrict__ attn) {
  __shared__ __align__(16) float sP[4][16 * 32];
  const int lane = threadIdx.x & 31;
  const int wave = threadIdx.x >> 5;
  const int head = blockIdx.x;
  const int kvh  = head / GG;
  const int m0   = blockIdx.y * 64 + wave * 16;
  const int r    = lane & 15;
  const int half = lane >> 4;

  const float* qrow = qkv + (size_t)(m0 + r) * QKVN + head * HDM;
  const v16h qa0 = frag_from_f32(qrow, 0, lane);
  const v16h qa1 = frag_from_f32(qrow, 32, lane);

  const _Float16* Kb = Kf + (size_t)kvh * TT * HDM;   // [t][d]
  const _Float16* Vb = Vf + (size_t)kvh * HDM * TT;   // [d][t]
  float* sp = sP[wave];

  v8f o0 = {}, o1 = {}, o2 = {}, o3 = {};
  float m_run[8], l_run[8];
#pragma unroll
  for (int i = 0; i < 8; ++i) { m_run[i] = -3.0e38f; l_run[i] = 0.f; }

  for (int t0 = 0; t0 < TT; t0 += 32) {
    // S tile (16 x 32) = Q @ K^T
    v8f s0 = {}, s1 = {};
    {
      const _Float16* k0row = Kb + (size_t)(t0 + r) * HDM;
      const _Float16* k1row = Kb + (size_t)(t0 + 16 + r) * HDM;
      v16h b;
      b = frag_from_f16(k0row, 0, lane);  s0 = wmma_f16(qa0, b, s0);
      b = frag_from_f16(k0row, 32, lane); s0 = wmma_f16(qa1, b, s0);
      b = frag_from_f16(k1row, 0, lane);  s1 = wmma_f16(qa0, b, s1);
      b = frag_from_f16(k1row, 32, lane); s1 = wmma_f16(qa1, b, s1);
    }
    // mask + online softmax (per C-layout row)
    const int c0 = t0 + r;
    const int c1 = c0 + 16;
#pragma unroll
    for (int i = 0; i < 8; ++i) {
      const int mrow = m0 + i + half * 8;
      const int lim  = PASTT + mrow;
      float v0 = s0[i] + ((c0 <= lim) ? 0.f : -1000000000.0f);
      float v1 = s1[i] + ((c1 <= lim) ? 0.f : -1000000000.0f);
      float mx = fmaxf(v0, v1);
      mx = fmaxf(mx, __shfl_xor(mx, 1, 32));
      mx = fmaxf(mx, __shfl_xor(mx, 2, 32));
      mx = fmaxf(mx, __shfl_xor(mx, 4, 32));
      mx = fmaxf(mx, __shfl_xor(mx, 8, 32));
      mx = fmaxf(mx, m_run[i]);
      const float scale = expf(m_run[i] - mx);
      m_run[i] = mx;
      const float p0 = expf(v0 - mx);
      const float p1 = expf(v1 - mx);
      float rs = p0 + p1;
      rs += __shfl_xor(rs, 1, 32);
      rs += __shfl_xor(rs, 2, 32);
      rs += __shfl_xor(rs, 4, 32);
      rs += __shfl_xor(rs, 8, 32);
      l_run[i] = l_run[i] * scale + rs;
      o0[i] *= scale; o1[i] *= scale; o2[i] *= scale; o3[i] *= scale;
      const int prow = i + half * 8;
      sp[prow * 32 + r]      = p0;
      sp[prow * 32 + 16 + r] = p1;
    }
    // O (16 x 64) += P (16 x 32) @ V (32 x 64)
    const v16h pa = frag_from_f32(sp + r * 32, 0, lane);
    const v16h b0 = frag_from_f16(Vb + (size_t)(0  + r) * TT, t0, lane);
    const v16h b1 = frag_from_f16(Vb + (size_t)(16 + r) * TT, t0, lane);
    const v16h b2 = frag_from_f16(Vb + (size_t)(32 + r) * TT, t0, lane);
    const v16h b3 = frag_from_f16(Vb + (size_t)(48 + r) * TT, t0, lane);
    o0 = wmma_f16(pa, b0, o0);
    o1 = wmma_f16(pa, b1, o1);
    o2 = wmma_f16(pa, b2, o2);
    o3 = wmma_f16(pa, b3, o3);
  }

#pragma unroll
  for (int i = 0; i < 8; ++i) {
    const float inv = 1.0f / l_run[i];
    const size_t off = (size_t)(m0 + i + half * 8) * HH + head * HDM + r;
    attn[off]      = o0[i] * inv;
    attn[off + 16] = o1[i] * inv;
    attn[off + 32] = o2[i] * inv;
    attn[off + 48] = o3[i] * inv;
  }
}

// act = silu(gate) * up
__global__ void silu_mul_kernel(const float* __restrict__ gu,
                                float* __restrict__ act) {
  int idx = blockIdx.x * blockDim.x + threadIdx.x;
  if (idx >= SS * DFFD) return;
  const int j = idx % DFFD;
  const int s = idx / DFFD;
  const float g = gu[(size_t)s * 2 * DFFD + j];
  const float u = gu[(size_t)s * 2 * DFFD + DFFD + j];
  act[idx] = g / (1.f + expf(-g)) * u;
}

__global__ void copy_f32_kernel(const float* __restrict__ src,
                                float* __restrict__ dst, int n) {
  int idx = blockIdx.x * blockDim.x + threadIdx.x;
  if (idx < n) dst[idx] = src[idx];
}

// ---------------------------------------------------------------------------
extern "C" void kernel_launch(void* const* d_in, const int* in_sizes, int n_in,
                              void* d_out, int out_size, void* d_ws, size_t ws_size,
                              hipStream_t stream) {
  const float* hs    = (const float*)d_in[0];
  const float* cosb  = (const float*)d_in[1];
  const float* sinb  = (const float*)d_in[2];
  /* d_in[3] attention_mask: computed inline (identical causal form) */
  const float* pastK = (const float*)d_in[4];
  const float* pastV = (const float*)d_in[5];
  const float* qkvW  = (const float*)d_in[6];
  const float* oW    = (const float*)d_in[7];
  const float* guW   = (const float*)d_in[8];
  const float* dW    = (const float*)d_in[9];
  const float* normW = (const float*)d_in[10];
  float* out = (float*)d_out;

  // workspace layout
  float* ws   = (float*)d_ws;
  float* h    = ws;                              // 256*1024
  float* xn   = h    + (size_t)SS * HH;          // 256*1024
  float* qkvb = xn   + (size_t)SS * HH;          // 256*1280
  float* attn = qkvb + (size_t)SS * QKVN;        // 256*1024
  float* gub  = attn + (size_t)SS * HH;          // 256*8192
  float* actb = gub  + (size_t)SS * 2 * DFFD;    // 256*4096
  _Float16* Kf = (_Float16*)(actb + (size_t)SS * DFFD); // 2*2304*64 f16
  _Float16* Vf = Kf + (size_t)KVHD * TT * HDM;          // 2*64*2304 f16

  copy_f32_kernel<<<(SS * HH + 255) / 256, 256, 0, stream>>>(hs, h, SS * HH);

  for (int l = 0; l < LL; ++l) {
    // pre-attention norm + QKV projection
    rmsnorm_sum_kernel<<<SS, 256, 0, stream>>>(h, xn);
    gemm_xwT_wmma<false><<<dim3(SS / 16, QKVN / 128), 128, 0, stream>>>(
        xn, qkvW + (size_t)l * QKVN * HH, qkvb, nullptr, QKVN, HH);

    // RoPE + f16 KV caches in WMMA-friendly layouts
    rope_qk_kernel<<<(SS * QKHD * 32 + 255) / 256, 256, 0, stream>>>(
        qkvb, cosb, sinb);
    build_K_kernel<<<(KVHD * TT * HDM + 255) / 256, 256, 0, stream>>>(
        qkvb, pastK + (size_t)l * KVHD * HDM * PASTT, Kf);
    build_V_kernel<<<(KVHD * HDM * TT + 255) / 256, 256, 0, stream>>>(
        qkvb, pastV + (size_t)l * KVHD * PASTT * HDM, Vf);

    // WMMA flash attention + output projection (residual fused)
    attn_wmma_kernel<<<dim3(NHD, SS / 64), 128, 0, stream>>>(qkvb, Kf, Vf, attn);
    gemm_xwT_wmma<true><<<dim3(SS / 16, HH / 128), 128, 0, stream>>>(
        attn, oW + (size_t)l * HH * HH, h, h, HH, HH);

    // MLP (residual fused into down-proj)
    rmsnorm_sum_kernel<<<SS, 256, 0, stream>>>(h, xn);
    gemm_xwT_wmma<false><<<dim3(SS / 16, (2 * DFFD) / 128), 128, 0, stream>>>(
        xn, guW + (size_t)l * 2 * DFFD * HH, gub, nullptr, 2 * DFFD, HH);
    silu_mul_kernel<<<(SS * DFFD + 255) / 256, 256, 0, stream>>>(gub, actb);
    gemm_xwT_wmma<true><<<dim3(SS / 16, HH / 128), 128, 0, stream>>>(
        actb, dW + (size_t)l * HH * DFFD, h, h, HH, DFFD);
  }

  final_norm_kernel<<<SS, 256, 0, stream>>>(h, normW, out);
}
